// AttentionDecoder_28114855920149
// MI455X (gfx1250) — matrix operand order repaired and checked by reading, not compiled, not run on hardware
//
#include <hip/hip_runtime.h>
#include <hip/hip_bf16.h>

// ---------------------------------------------------------------------------
// AttentionDecoder for MI455X (gfx1250).
// All GEMMs on v_wmma_f32_16x16x32_bf16, f32 accumulation.
// - Weights pre-packed to bf16 (N-major) once: inner loops are pure
//   {b128 loads, wmma} streams, software-pipelined (load t+1 under wmma t).
// - Fused attention kernel: keys chunks are DMA'd into LDS with
//   global_load_async_to_lds_b128 (ASYNCcnt) one chunk ahead of compute,
//   converted once per block to bf16, consumed by 8 waves x 8 WMMAs.
// Shapes: V=32000 E=512 H=1024 K=2048 L=196 B=256
// ---------------------------------------------------------------------------

#define VSZ 32000
#define ESZ 512
#define HSZ 1024
#define KSZ 2048
#define LSZ 196
#define BSZ 256

#define ACHUNK 256                 // k-chunk of keys staged in LDS
#define ASTR   (ACHUNK + 8)        // padded bf16 row stride: distinct banks

typedef __attribute__((ext_vector_type(16))) __bf16 v16bf;
typedef __attribute__((ext_vector_type(8)))  __bf16 v8bf;
typedef __attribute__((ext_vector_type(8)))  float  v8f;

union bfvec { v16bf v; v8bf h[2]; };

__device__ __forceinline__ __bf16 f2bf(float f) { return static_cast<__bf16>(f); }

__device__ __forceinline__ float sigmoidf_(float x) {
    return 1.0f / (1.0f + __expf(-x));
}

// LDS byte offset of a generic pointer to shared memory (ISA: addr[31:0]).
__device__ __forceinline__ unsigned lds_off(const void* p) {
    return (unsigned)(uintptr_t)p;
}

// Async DMA: 16 bytes global -> LDS, tracked by ASYNCcnt.
__device__ __forceinline__ void async_g2l_b128(unsigned ldsByteOff, const void* gaddr) {
    asm volatile("global_load_async_to_lds_b128 %0, %1, off"
                 :: "v"(ldsByteOff), "v"(gaddr) : "memory");
}
__device__ __forceinline__ void wait_asynccnt0() {
    asm volatile("s_wait_asynccnt 0x0" ::: "memory");
}

// ---------------------------------------------------------------------------
// Pack f32 -> bf16, same layout.
// ---------------------------------------------------------------------------
__global__ void pack_bf16_kernel(const float* __restrict__ in,
                                 __bf16* __restrict__ out, long n2)
{
    const long i = (long)blockIdx.x * blockDim.x + threadIdx.x;
    if (i >= n2) return;
    out[2 * i]     = f2bf(in[2 * i]);
    out[2 * i + 1] = f2bf(in[2 * i + 1]);
}

// ---------------------------------------------------------------------------
// Transpose + pack: in is Kd x N (f32, row-major) -> out is N x Kd (bf16).
// ---------------------------------------------------------------------------
__global__ void transpose_bf16_kernel(const float* __restrict__ in,
                                      __bf16* __restrict__ out, int Kd, int N)
{
    __shared__ float tile[16][17];
    const int kIn = blockIdx.y * 16 + threadIdx.y;
    const int nIn = blockIdx.x * 16 + threadIdx.x;
    tile[threadIdx.y][threadIdx.x] = in[(long)kIn * N + nIn];
    __syncthreads();
    const int nOut = blockIdx.x * 16 + threadIdx.y;
    const int kOut = blockIdx.y * 16 + threadIdx.x;
    out[(long)nOut * Kd + kOut] = f2bf(tile[threadIdx.x][threadIdx.y]);
}

// ---------------------------------------------------------------------------
// bf16 WMMA GEMM:  C[M,N] = A[M,Kd](bf16) * Bt[N,Kd](bf16)^T + bias
// One wave owns a 16x64 tile (4 accumulators), software-pipelined loads.
// ---------------------------------------------------------------------------
__global__ void gemm_bf16_wmma(const __bf16* __restrict__ A,
                               const __bf16* __restrict__ Bt,
                               const float* __restrict__ bias,
                               float* __restrict__ C,
                               int M, int N, int Kd,
                               int lda, int ldb, int ldc, int cColOff)
{
    const int gwave = (blockIdx.x * blockDim.x + threadIdx.x) >> 5;
    const int lane  = threadIdx.x & 31;
    const int ngrp  = N >> 6;
    const int mtile = gwave / ngrp;
    const int ng    = gwave % ngrp;
    if (mtile * 16 >= M) return;

    const int mn   = lane & 15;
    const int half = lane >> 4;
    const int kA   = half << 4;
    const int kB   = half << 4;

    const __bf16* Arow  = A  + (long)(mtile * 16 + mn) * lda;
    const __bf16* Brow0 = Bt + (long)(ng * 64 + mn) * ldb;

    auto loadA = [&](int k) {
        bfvec x;
        x.h[0] = *(const v8bf*)(Arow + k + kA);
        x.h[1] = *(const v8bf*)(Arow + k + kA + 8);
        return x;
    };
    auto loadB = [&](int k, int t) {
        const __bf16* p = Brow0 + (long)t * 16 * ldb + k + kB;
        bfvec x;
        x.h[0] = *(const v8bf*)(p);
        x.h[1] = *(const v8bf*)(p + 8);
        return x;
    };

    v8f acc[4] = {v8f{}, v8f{}, v8f{}, v8f{}};

    bfvec aCur = loadA(0);
    bfvec bCur = loadB(0, 0);
    for (int k0 = 0; k0 < Kd; k0 += 32) {
        const int kN = (k0 + 32 < Kd) ? k0 + 32 : k0;
        __builtin_prefetch(Arow + kN + 32, 0, 1);
        bfvec aNext;
#pragma unroll
        for (int t = 0; t < 4; ++t) {
            bfvec bNext;
            if (t < 3) {
                bNext = loadB(k0, t + 1);
            } else {
                aNext = loadA(kN);
                bNext = loadB(kN, 0);
            }
            acc[t] = __builtin_amdgcn_wmma_f32_16x16x32_bf16(
                         false, aCur.v, false, bCur.v, (short)0, acc[t], false, false);
            bCur = bNext;
        }
        aCur = aNext;
    }

    const int mbase = half << 3;
#pragma unroll
    for (int t = 0; t < 4; ++t) {
        const int n  = ng * 64 + t * 16 + mn;
        const float bs = bias ? bias[n] : 0.0f;
#pragma unroll
        for (int r = 0; r < 8; ++r) {
            const int mm = mtile * 16 + mbase + r;
            C[(long)mm * ldc + cColOff + n] = acc[t][r] + bs;
        }
    }
}

// ---------------------------------------------------------------------------
// Fused attention scores:
//   scores[b,l] = sum_h tanh(q[b,h] + bk[h] + sum_k keys[b,l,k]*Wk[k,h]) * v[h] + bv
// Block = (b, l-tile of 16), 8 waves; wave w owns h in [w*128,(w+1)*128).
// keys chunks (16 x 256 f32) DMA'd to LDS one chunk ahead via
// global_load_async_to_lds_b128; converted to bf16 in LDS once per block;
// B tiles (WkT, L2-resident) pipelined in registers. kproj never hits memory.
// ---------------------------------------------------------------------------
__global__ void attn_scores_kernel(const float* __restrict__ keys,
                                   const __bf16* __restrict__ WkT,   // H x K bf16
                                   const float* __restrict__ q,
                                   const float* __restrict__ bk,
                                   const float* __restrict__ vvec,
                                   const float* __restrict__ bv,
                                   float* __restrict__ scores)
{
    __shared__ __align__(16) float  fslab[2][16 * ACHUNK];   // 2 x 16 KB f32 staging
    __shared__ __align__(16) __bf16 aslab[16 * ASTR];        // ~8.3 KB bf16
    __shared__ float red[8][32][8];                          // 8 KB

    const int ltile = blockIdx.x;
    const int b     = blockIdx.y;
    const int tid   = threadIdx.x;
    const int wave  = tid >> 5;
    const int lane  = tid & 31;
    const int mn    = lane & 15;
    const int half  = lane >> 4;
    const int kA    = half << 4;
    const int kB    = half << 4;

    // per-thread DMA / conversion mapping: (row, 16-col span)
    const int crow = tid >> 4;              // 0..15
    const int ccol = (tid & 15) << 4;       // 0,16,...,240
    int lsrc = ltile * 16 + crow; if (lsrc > LSZ - 1) lsrc = LSZ - 1;
    const float* krow = keys + ((long)b * LSZ + lsrc) * KSZ;

    // issue the async DMA of one 16x256 chunk into fslab[buf]
    auto dma_chunk = [&](int kc, int buf) {
#pragma unroll
        for (int j = 0; j < 16; j += 4) {
            async_g2l_b128(lds_off(&fslab[buf][crow * ACHUNK + ccol + j]),
                           krow + kc + ccol + j);
        }
    };

    const __bf16* Brow0 = WkT + (long)(wave * 128 + mn) * KSZ;
    auto loadB = [&](int k, int t) {
        const __bf16* p = Brow0 + (long)t * 16 * KSZ + k + kB;
        bfvec x;
        x.h[0] = *(const v8bf*)(p);
        x.h[1] = *(const v8bf*)(p + 8);
        return x;
    };

    v8f acc[8] = {v8f{}, v8f{}, v8f{}, v8f{}, v8f{}, v8f{}, v8f{}, v8f{}};
    bfvec bCur = loadB(0, 0);

    dma_chunk(0, 0);                      // prologue: chunk 0 in flight
    int buf = 0;

    for (int kc = 0; kc < KSZ; kc += ACHUNK) {
        wait_asynccnt0();                 // chunk kc landed in fslab[buf]
        __syncthreads();                  // all waves done with previous aslab

        if (kc + ACHUNK < KSZ)            // DMA next chunk under this chunk's WMMAs
            dma_chunk(kc + ACHUNK, buf ^ 1);

        // cooperative f32 -> bf16 conversion from LDS staging
#pragma unroll
        for (int j = 0; j < 16; j += 4) {
            const float4 f = *(const float4*)(&fslab[buf][crow * ACHUNK + ccol + j]);
            aslab[crow * ASTR + ccol + j + 0] = f2bf(f.x);
            aslab[crow * ASTR + ccol + j + 1] = f2bf(f.y);
            aslab[crow * ASTR + ccol + j + 2] = f2bf(f.z);
            aslab[crow * ASTR + ccol + j + 3] = f2bf(f.w);
        }
        __syncthreads();                  // bf16 chunk visible to all waves

        for (int ks = 0; ks < ACHUNK; ks += 32) {
            const int k    = kc + ks;
            const int kNxt = (k + 32 < KSZ) ? k + 32 : k;
            bfvec a;
            const __bf16* ap = &aslab[mn * ASTR + ks + kA];
            a.h[0] = *(const v8bf*)(ap);
            a.h[1] = *(const v8bf*)(ap + 8);
#pragma unroll
            for (int t = 0; t < 8; ++t) {
                const bfvec bNext = (t < 7) ? loadB(k, t + 1) : loadB(kNxt, 0);
                acc[t] = __builtin_amdgcn_wmma_f32_16x16x32_bf16(
                             false, a.v, false, bCur.v, (short)0, acc[t], false, false);
                bCur = bNext;
            }
        }
        buf ^= 1;
    }

    float partial[8];
#pragma unroll
    for (int r = 0; r < 8; ++r) partial[r] = 0.0f;

#pragma unroll
    for (int t = 0; t < 8; ++t) {
        const int h    = wave * 128 + t * 16 + mn;
        const float qv = q[(long)b * HSZ + h] + bk[h];
        const float vh = vvec[h];
#pragma unroll
        for (int r = 0; r < 8; ++r)
            partial[r] += tanhf(acc[t][r] + qv) * vh;
    }

#pragma unroll
    for (int r = 0; r < 8; ++r) red[wave][lane][r] = partial[r];
    __syncthreads();

    if (threadIdx.x < 16) {
        const int m    = threadIdx.x;
        const int lnof = (m >= 8) ? 16 : 0;
        const int r    = m & 7;
        float s = 0.0f;
        for (int w = 0; w < 8; ++w)
            for (int ln = 0; ln < 16; ++ln)
                s += red[w][lnof + ln][r];
        const int l = ltile * 16 + m;
        if (l < LSZ) scores[(long)b * LSZ + l] = s + bv[0];
    }
}

// ---------------------------------------------------------------------------
// Softmax over L + context = sum_l weights[l]*keys[b,l,:]. One block per b.
// ---------------------------------------------------------------------------
__global__ void softmax_context_kernel(const float* __restrict__ scores,
                                       const float* __restrict__ keys,
                                       float* __restrict__ weights_out,
                                       float* __restrict__ context)
{
    __shared__ float sw[LSZ];
    __shared__ float smax, ssum;
    const int b   = blockIdx.x;
    const int tid = threadIdx.x;

    for (int l = tid; l < LSZ; l += blockDim.x)
        sw[l] = scores[(long)b * LSZ + l];
    __syncthreads();

    if (tid == 0) {
        float mx = sw[0];
        for (int l = 1; l < LSZ; ++l) mx = fmaxf(mx, sw[l]);
        smax = mx;
    }
    __syncthreads();
    if (tid == 0) {
        float s = 0.0f;
        for (int l = 0; l < LSZ; ++l) s += __expf(sw[l] - smax);
        ssum = s;
    }
    __syncthreads();

    for (int l = tid; l < LSZ; l += blockDim.x) {
        const float w = __expf(sw[l] - smax) / ssum;
        sw[l] = w;
        weights_out[(long)b * LSZ + l] = w;
    }
    __syncthreads();

    const float* kb = keys + (long)b * LSZ * KSZ;
    for (int k = tid; k < KSZ; k += blockDim.x) {
        float acc = 0.0f;
        for (int l = 0; l < LSZ; ++l)
            acc = fmaf(sw[l], kb[(long)l * KSZ + k], acc);
        context[(long)b * KSZ + k] = acc;
    }
}

// x = emb[inputs]; written into rnn_in columns [0, E)
__global__ void gather_embed_kernel(const int* __restrict__ inputs,
                                    const float* __restrict__ emb,
                                    float* __restrict__ rnn_in)
{
    const int i = blockIdx.x * blockDim.x + threadIdx.x;
    if (i >= BSZ * ESZ) return;
    const int b = i / ESZ, e = i % ESZ;
    rnn_in[(long)b * (ESZ + HSZ) + e] = emb[(long)inputs[b] * ESZ + e];
}

// GRU gate fusion -> h_new
__global__ void gru_kernel(const float* __restrict__ gi,
                           const float* __restrict__ gh,
                           const float* __restrict__ hidden,
                           float* __restrict__ h_new)
{
    const int i = blockIdx.x * blockDim.x + threadIdx.x;
    if (i >= BSZ * HSZ) return;
    const int b = i / HSZ, h = i % HSZ;
    const long base = (long)b * 3 * HSZ;
    const float ir = gi[base + h],            hr = gh[base + h];
    const float iz = gi[base + HSZ + h],      hz = gh[base + HSZ + h];
    const float in_ = gi[base + 2 * HSZ + h], hn = gh[base + 2 * HSZ + h];
    const float r = sigmoidf_(ir + hr);
    const float z = sigmoidf_(iz + hz);
    const float n = tanhf(in_ + r * hn);
    h_new[i] = (1.0f - z) * n + z * hidden[i];
}

// ---------------------------------------------------------------------------
extern "C" void kernel_launch(void* const* d_in, const int* in_sizes, int n_in,
                              void* d_out, int out_size, void* d_ws, size_t ws_size,
                              hipStream_t stream)
{
    (void)in_sizes; (void)n_in; (void)out_size; (void)ws_size;
    const int*   inputs = (const int*)  d_in[0];
    const float* hidden = (const float*)d_in[1];
    const float* keys   = (const float*)d_in[2];
    const float* emb    = (const float*)d_in[3];
    const float* Wq     = (const float*)d_in[4];
    const float* bq     = (const float*)d_in[5];
    const float* Wk     = (const float*)d_in[6];
    const float* bk     = (const float*)d_in[7];
    const float* vvec   = (const float*)d_in[8];
    const float* bv     = (const float*)d_in[9];
    const float* fc0_W  = (const float*)d_in[10];
    const float* fc0_b  = (const float*)d_in[11];
    const float* W_ih   = (const float*)d_in[12];
    const float* W_hh   = (const float*)d_in[13];
    const float* b_ih   = (const float*)d_in[14];
    const float* b_hh   = (const float*)d_in[15];
    const float* fc1_W  = (const float*)d_in[16];
    const float* fc1_b  = (const float*)d_in[17];

    float* logits  = (float*)d_out;                       // B x V
    float* h_new   = logits + (long)BSZ * VSZ;            // B x H
    float* weights = h_new + (long)BSZ * HSZ;             // B x L

    // ---- workspace: f32 region first, then bf16 region ----
    float* wsf     = (float*)d_ws;
    float* q       = wsf;  wsf += (long)BSZ * HSZ;
    float* scores  = wsf;  wsf += (long)BSZ * LSZ;
    float* context = wsf;  wsf += (long)BSZ * KSZ;
    float* rnn_in  = wsf;  wsf += (long)BSZ * (ESZ + HSZ);
    float* gi      = wsf;  wsf += (long)BSZ * 3 * HSZ;
    float* gh      = wsf;  wsf += (long)BSZ * 3 * HSZ;

    __bf16* wsb    = (__bf16*)wsf;
    __bf16* hid_bf = wsb;  wsb += (long)BSZ * HSZ;
    __bf16* WqT    = wsb;  wsb += (long)HSZ * HSZ;
    __bf16* WkT    = wsb;  wsb += (long)HSZ * KSZ;
    __bf16* fc0T   = wsb;  wsb += (long)HSZ * KSZ;
    __bf16* fc1T   = wsb;  wsb += (long)VSZ * HSZ;
    __bf16* Wih_bf = wsb;  wsb += (long)3 * HSZ * (ESZ + HSZ);
    __bf16* Whh_bf = wsb;  wsb += (long)3 * HSZ * HSZ;
    __bf16* ctx_bf = wsb;  wsb += (long)BSZ * KSZ;
    __bf16* rnn_bf = wsb;  wsb += (long)BSZ * (ESZ + HSZ);
    __bf16* hn_bf  = wsb;  wsb += (long)BSZ * HSZ;

    const int TPB = 256;   // 8 wave32s
    auto packN  = [&](const float* src, __bf16* dst, long n) {
        pack_bf16_kernel<<<(unsigned)((n / 2 + TPB - 1) / TPB), TPB, 0, stream>>>(src, dst, n / 2);
    };
    auto transN = [&](const float* src, __bf16* dst, int Kd, int N) {
        transpose_bf16_kernel<<<dim3(N / 16, Kd / 16), dim3(16, 16), 0, stream>>>(src, dst, Kd, N);
    };
    auto gemm_blocks = [](int M, int N) { return ((M / 16) * (N / 64)) / 8; };

    // 0) one-time bf16 packing of weights and hidden
    packN (hidden, hid_bf, (long)BSZ * HSZ);
    transN(Wq,    WqT,  HSZ, HSZ);
    transN(Wk,    WkT,  KSZ, HSZ);
    transN(fc0_W, fc0T, KSZ, HSZ);
    transN(fc1_W, fc1T, HSZ, VSZ);
    packN (W_ih,  Wih_bf, (long)3 * HSZ * (ESZ + HSZ));
    packN (W_hh,  Whh_bf, (long)3 * HSZ * HSZ);

    // 1) x = emb[inputs] -> rnn_in[:, 0:E]
    gather_embed_kernel<<<(BSZ * ESZ + TPB - 1) / TPB, TPB, 0, stream>>>(inputs, emb, rnn_in);

    // 2) q = hidden @ Wq + bq
    gemm_bf16_wmma<<<gemm_blocks(BSZ, HSZ), TPB, 0, stream>>>(
        hid_bf, WqT, bq, q, BSZ, HSZ, HSZ, HSZ, HSZ, HSZ, 0);

    // 3) fused scores (the 210-GFLOP stage)
    attn_scores_kernel<<<dim3((LSZ + 15) / 16, BSZ), TPB, 0, stream>>>(
        keys, WkT, q, bk, vvec, bv, scores);

    // 4) softmax -> weights (output), context
    softmax_context_kernel<<<BSZ, TPB, 0, stream>>>(scores, keys, weights, context);

    // 5) rnn_in[:, E:] = context @ fc0_W + fc0_b
    packN(context, ctx_bf, (long)BSZ * KSZ);
    gemm_bf16_wmma<<<gemm_blocks(BSZ, HSZ), TPB, 0, stream>>>(
        ctx_bf, fc0T, fc0_b, rnn_in, BSZ, HSZ, KSZ, KSZ, KSZ, ESZ + HSZ, ESZ);

    // 6) gi = rnn_in @ W_ih^T + b_ih
    packN(rnn_in, rnn_bf, (long)BSZ * (ESZ + HSZ));
    gemm_bf16_wmma<<<gemm_blocks(BSZ, 3 * HSZ), TPB, 0, stream>>>(
        rnn_bf, Wih_bf, b_ih, gi, BSZ, 3 * HSZ, ESZ + HSZ, ESZ + HSZ, ESZ + HSZ, 3 * HSZ, 0);

    // 7) gh = hidden @ W_hh^T + b_hh
    gemm_bf16_wmma<<<gemm_blocks(BSZ, 3 * HSZ), TPB, 0, stream>>>(
        hid_bf, Whh_bf, b_hh, gh, BSZ, 3 * HSZ, HSZ, HSZ, HSZ, 3 * HSZ, 0);

    // 8) GRU -> h_new (output)
    gru_kernel<<<(BSZ * HSZ + TPB - 1) / TPB, TPB, 0, stream>>>(gi, gh, hidden, h_new);

    // 9) logits = h_new @ fc1_W + fc1_b
    packN(h_new, hn_bf, (long)BSZ * HSZ);
    gemm_bf16_wmma<<<gemm_blocks(BSZ, VSZ), TPB, 0, stream>>>(
        hn_bf, fc1T, fc1_b, logits, BSZ, VSZ, HSZ, HSZ, HSZ, VSZ, 0);
}
